// SelfAttention2D_18382460027120
// MI455X (gfx1250) — compile-verified, matrix operand order checked by reading, not run on hardware
//
#include <hip/hip_runtime.h>
#include <hip/hip_bf16.h>
#include <math.h>

// ---------------------------------------------------------------------------
// SelfAttention2D on MI455X (gfx1250): all four GEMMs on v_wmma_f32_16x16x32_f16.
// B=4, C_in=256, d=64, N=64*64=4096. Flash-attention fusion (no N x N buffer).
// ---------------------------------------------------------------------------

typedef __attribute__((ext_vector_type(16))) _Float16 v16h;
typedef __attribute__((ext_vector_type(8)))  _Float16 v8h;
typedef __attribute__((ext_vector_type(8)))  float    v8f;

#define WMMA_F16(A, B, C) \
    __builtin_amdgcn_wmma_f32_16x16x32_f16(false, (A), false, (B), (short)0, (C), false, false)

__device__ __forceinline__ v16h cat16(v8h a, v8h b) {
    return __builtin_shufflevector(a, b, 0,1,2,3,4,5,6,7,8,9,10,11,12,13,14,15);
}

// Max-reduce across the 16-lane DPP row (our C-layout row group) with
// row_ror rotations: pure VALU, no LDS pipe traffic.
__device__ __forceinline__ float rowmax16(float x) {
    int t;
    t = __builtin_amdgcn_update_dpp(__builtin_bit_cast(int, x), __builtin_bit_cast(int, x),
                                    0x121, 0xf, 0xf, false);            // row_ror:1
    x = fmaxf(x, __builtin_bit_cast(float, t));
    t = __builtin_amdgcn_update_dpp(__builtin_bit_cast(int, x), __builtin_bit_cast(int, x),
                                    0x122, 0xf, 0xf, false);            // row_ror:2
    x = fmaxf(x, __builtin_bit_cast(float, t));
    t = __builtin_amdgcn_update_dpp(__builtin_bit_cast(int, x), __builtin_bit_cast(int, x),
                                    0x124, 0xf, 0xf, false);            // row_ror:4
    x = fmaxf(x, __builtin_bit_cast(float, t));
    t = __builtin_amdgcn_update_dpp(__builtin_bit_cast(int, x), __builtin_bit_cast(int, x),
                                    0x128, 0xf, 0xf, false);            // row_ror:8
    x = fmaxf(x, __builtin_bit_cast(float, t));
    return x;
}

static constexpr int N    = 4096;   // H*W
static constexpr int CIN  = 256;
static constexpr int DHEA = 64;

// ---------------------------------------------------------------------------
// Stage 0: convert the three projection weight matrices to f16 once.
// Wh layout: [3][64][256] (q, k, v)
// ---------------------------------------------------------------------------
__global__ void __launch_bounds__(256) wconv_kernel(
    const float* __restrict__ wq, const float* __restrict__ wk,
    const float* __restrict__ wv, _Float16* __restrict__ Wh)
{
    const int idx = blockIdx.x * 256 + threadIdx.x;   // 3*64*256 = 49152
    if (idx >= 3 * DHEA * CIN) return;
    const int m = idx >> 14, r = idx & 16383;         // 16384 per matrix
    const float* src = (m == 0) ? wq : (m == 1) ? wk : wv;
    Wh[idx] = (_Float16)src[r];
}

// ---------------------------------------------------------------------------
// Stage A: 1x1 conv projections q/k/v as WMMA. One wave per (batch, 16-n tile):
// x tile transposed to LDS f16, then 3 proj x 4 d-tiles x 8 k-chunks = 96 wmma.
//   Qt, Kt : [B][N][64]  (n-major, d contiguous)
//   Vd     : [B][64][N]  (d-major, n contiguous)
// ---------------------------------------------------------------------------
__global__ void __launch_bounds__(32) proj_kernel(
    const float* __restrict__ x, const _Float16* __restrict__ Wh,
    const float* __restrict__ bq, const float* __restrict__ bk,
    const float* __restrict__ bv,
    _Float16* __restrict__ Qt, _Float16* __restrict__ Kt,
    _Float16* __restrict__ Vd)
{
    const int b    = blockIdx.x >> 8;
    const int n0   = (blockIdx.x & 255) * 16;
    const int lane = threadIdx.x;
    const int h    = lane >> 4;
    const int M    = lane & 15;

    // x tile transposed: xt[n][c], f16, row padded to 264 (16B-aligned stride,
    // rows land on distinct banks for the b128 fragment reads).
    __shared__ _Float16 xt[16][264];
    for (int idx = lane; idx < CIN * 16; idx += 32) {
        const int c = idx >> 4, j = idx & 15;
        xt[j][c] = (_Float16)x[((size_t)(b * CIN + c)) * N + n0 + j];
    }
    __syncthreads();

    // B fragments (x): one per 32-wide k-chunk, reused by all 12 output tiles
    v16h xB[8];
    #pragma unroll
    for (int kc = 0; kc < 8; ++kc) {
        const _Float16* row = &xt[M][kc * 32];
        xB[kc] = cat16(*(const v8h*)(row + h * 8), *(const v8h*)(row + 16 + h * 8));
    }

    #pragma unroll
    for (int p = 0; p < 3; ++p) {
        const _Float16* wp = Wh + (size_t)p * DHEA * CIN;
        const float*  bias = (p == 0) ? bq : (p == 1) ? bk : bv;
        #pragma unroll
        for (int dt = 0; dt < 4; ++dt) {
            v8f acc;
            #pragma unroll
            for (int r = 0; r < 8; ++r) acc[r] = bias[dt * 16 + h * 8 + r];
            #pragma unroll
            for (int kc = 0; kc < 8; ++kc) {
                const _Float16* wrow = wp + (size_t)(dt * 16 + M) * CIN + kc * 32;
                v16h aW = cat16(*(const v8h*)(wrow + h * 8),
                                *(const v8h*)(wrow + 16 + h * 8));
                acc = WMMA_F16(aW, xB[kc], acc);
            }
            #pragma unroll
            for (int r = 0; r < 8; ++r) {
                const int d = dt * 16 + h * 8 + r;
                const _Float16 v = (_Float16)acc[r];
                if (p == 0)      Qt[(size_t)(b * N + n0 + M) * DHEA + d] = v;
                else if (p == 1) Kt[(size_t)(b * N + n0 + M) * DHEA + d] = v;
                else             Vd[(size_t)(b * DHEA + d) * N + n0 + M] = v;
            }
        }
    }
}

// ---------------------------------------------------------------------------
// Stage B: flash attention. One wave per (batch, 16-query block), 64-key tiles.
// 8 wmma for S, 8 for O, 2 for the row-sum (all-ones B fragment -> l in O[4]).
// Row max via DPP; P relaid to A-layout through 2.5 KB LDS.
// ---------------------------------------------------------------------------
__global__ void __launch_bounds__(32) attn_kernel(
    const _Float16* __restrict__ Qt, const _Float16* __restrict__ Kt,
    const _Float16* __restrict__ Vd, float* __restrict__ Obuf)
{
    const int b    = blockIdx.x >> 8;
    const int q0   = (blockIdx.x & 255) * 16;
    const int lane = threadIdx.x;
    const int h    = lane >> 4;
    const int M    = lane & 15;

    // ---- Q in A-layout, 2 chunks of 32 over d -----------------------------
    v16h qA[2];
    {
        const _Float16* qrow = Qt + (size_t)(b * N + q0 + M) * DHEA;
        #pragma unroll
        for (int c = 0; c < 2; ++c)
            qA[c] = cat16(*(const v8h*)(qrow + c * 32 + h * 8),
                          *(const v8h*)(qrow + c * 32 + 16 + h * 8));
    }

    // all-ones column-0 B fragment: accumulates row sums of P into O[4]
    v16h lB;
    {
        const _Float16 lv = (M == 0) ? (_Float16)1.0f : (_Float16)0.0f;
        #pragma unroll
        for (int e = 0; e < 16; ++e) lB[e] = lv;
    }

    float mrow[8];
    v8f O[5];
    #pragma unroll
    for (int r = 0; r < 8; ++r) mrow[r] = -INFINITY;
    #pragma unroll
    for (int c = 0; c < 5; ++c) O[c] = v8f{};

    __shared__ _Float16 lds_p[16][80];   // 64 cols + pad (16B-aligned stride)

    for (int kt = 0; kt < N; kt += 64) {
        // ---- S = Q^T K : four 16x16 tiles over 64 keys --------------------
        v8f S[4];
        #pragma unroll
        for (int t = 0; t < 4; ++t) {
            const _Float16* krow = Kt + (size_t)(b * N + kt + t * 16 + M) * DHEA;
            v16h k0 = cat16(*(const v8h*)(krow + h * 8),
                            *(const v8h*)(krow + 16 + h * 8));
            v16h k1 = cat16(*(const v8h*)(krow + 32 + h * 8),
                            *(const v8h*)(krow + 48 + h * 8));
            v8f s = v8f{};
            s = WMMA_F16(qA[0], k0, s);
            s = WMMA_F16(qA[1], k1, s);
            S[t] = s;
        }

        // ---- online softmax: DPP row max, exp, stash P in LDS -------------
        float corr[8];
        #pragma unroll
        for (int r = 0; r < 8; ++r) {
            float rm = fmaxf(fmaxf(S[0][r], S[1][r]), fmaxf(S[2][r], S[3][r]));
            rm = rowmax16(rm);
            const float nm = fmaxf(mrow[r], rm);
            corr[r] = __expf(mrow[r] - nm);
            mrow[r] = nm;
            #pragma unroll
            for (int t = 0; t < 4; ++t)
                lds_p[h * 8 + r][t * 16 + M] = (_Float16)__expf(S[t][r] - nm);
        }
        #pragma unroll
        for (int c = 0; c < 5; ++c)
            #pragma unroll
            for (int r = 0; r < 8; ++r) O[c][r] *= corr[r];

        __syncthreads();
        v16h pA0, pA1;
        {
            const _Float16* prow = &lds_p[M][0];
            pA0 = cat16(*(const v8h*)(prow + h * 8),
                        *(const v8h*)(prow + 16 + h * 8));
            pA1 = cat16(*(const v8h*)(prow + 32 + h * 8),
                        *(const v8h*)(prow + 48 + h * 8));
        }
        __syncthreads();

        // ---- O^T += P x V^T, plus l += P x ones ---------------------------
        #pragma unroll
        for (int c = 0; c < 4; ++c) {
            const _Float16* vrow = Vd + (size_t)(b * DHEA + c * 16 + M) * N + kt;
            v16h v0 = cat16(*(const v8h*)(vrow + h * 8),
                            *(const v8h*)(vrow + 16 + h * 8));
            v16h v1 = cat16(*(const v8h*)(vrow + 32 + h * 8),
                            *(const v8h*)(vrow + 48 + h * 8));
            O[c] = WMMA_F16(pA0, v0, O[c]);
            O[c] = WMMA_F16(pA1, v1, O[c]);
        }
        O[4] = WMMA_F16(pA0, lB, O[4]);
        O[4] = WMMA_F16(pA1, lB, O[4]);
    }

    // ---- normalize by l (column 0 of O[4]) and store O^T [N][64] ----------
    #pragma unroll
    for (int r = 0; r < 8; ++r) {
        const float lv  = __shfl(O[4][r], lane & 16, 32);  // lane h*16 holds l
        const float inv = 1.0f / lv;
        #pragma unroll
        for (int c = 0; c < 4; ++c)
            Obuf[(size_t)(b * N + q0 + h * 8 + r) * DHEA + c * 16 + M] = O[c][r] * inv;
    }
}

// ---------------------------------------------------------------------------
// Stage C: y = w_u @ O + b_u + x  (WMMA, one wave per 16c x 16n tile)
// ---------------------------------------------------------------------------
__global__ void __launch_bounds__(32) out_kernel(
    const float* __restrict__ x, const float* __restrict__ wu,
    const float* __restrict__ bu, const float* __restrict__ Obuf,
    float* __restrict__ y)
{
    const int nt   = blockIdx.x & 255;
    const int ct   = (blockIdx.x >> 8) & 15;
    const int b    = blockIdx.x >> 12;
    const int n0   = nt * 16;
    const int lane = threadIdx.x;
    const int h    = lane >> 4;
    const int M    = lane & 15;

    v16h aW[2], bO[2];
    {
        const float* wrow = wu + (size_t)(ct * 16 + M) * DHEA;
        const float* orow = Obuf + (size_t)(b * N + n0 + M) * DHEA;
        #pragma unroll
        for (int c = 0; c < 2; ++c) {
            v16h tw, to;
            #pragma unroll
            for (int e = 0; e < 8; ++e) {
                tw[e]     = (_Float16)wrow[c * 32 + h * 8 + e];
                tw[e + 8] = (_Float16)wrow[c * 32 + 16 + h * 8 + e];
                to[e]     = (_Float16)orow[c * 32 + h * 8 + e];
                to[e + 8] = (_Float16)orow[c * 32 + 16 + h * 8 + e];
            }
            aW[c] = tw;
            bO[c] = to;
        }
    }
    v8f acc = v8f{};
    acc = WMMA_F16(aW[0], bO[0], acc);
    acc = WMMA_F16(aW[1], bO[1], acc);

    #pragma unroll
    for (int r = 0; r < 8; ++r) {
        const int c = ct * 16 + h * 8 + r;
        const size_t idx = (size_t)(b * CIN + c) * N + n0 + M;
        y[idx] = acc[r] + bu[c] + x[idx];
    }
}

// ---------------------------------------------------------------------------
extern "C" void kernel_launch(void* const* d_in, const int* in_sizes, int n_in,
                              void* d_out, int out_size, void* d_ws, size_t ws_size,
                              hipStream_t stream)
{
    (void)in_sizes; (void)n_in; (void)out_size; (void)ws_size;
    const float* x  = (const float*)d_in[0];
    const float* wk = (const float*)d_in[1];
    const float* bk = (const float*)d_in[2];
    const float* wq = (const float*)d_in[3];
    const float* bq = (const float*)d_in[4];
    const float* wv = (const float*)d_in[5];
    const float* bv = (const float*)d_in[6];
    const float* wu = (const float*)d_in[7];
    const float* bu = (const float*)d_in[8];
    float* y = (float*)d_out;

    char* ws = (char*)d_ws;
    _Float16* Qt   = (_Float16*)(ws);                        // 2 MB
    _Float16* Kt   = (_Float16*)(ws + (2u << 20));           // 2 MB
    _Float16* Vd   = (_Float16*)(ws + (4u << 20));           // 2 MB
    float*    Obuf = (float*)   (ws + (6u << 20));           // 4 MB
    _Float16* Wh   = (_Float16*)(ws + (10u << 20));          // 96 KB f16 weights

    wconv_kernel<<<(3 * 64 * 256 + 255) / 256, 256, 0, stream>>>(wq, wk, wv, Wh);
    proj_kernel<<<4 * 256, 32, 0, stream>>>(x, Wh, bq, bk, bv, Qt, Kt, Vd);
    attn_kernel<<<4 * 256, 32, 0, stream>>>(Qt, Kt, Vd, Obuf);
    out_kernel<<<4 * 16 * 256, 32, 0, stream>>>(x, wu, bu, Obuf, y);
}
